// HaloAttention_65257733096102
// MI455X (gfx1250) — matrix operand order repaired and checked by reading, not compiled, not run on hardware
//
#include <hip/hip_runtime.h>
#include <hip/hip_bf16.h>
#include <math.h>

// ---------------------------------------------------------------------------
// Problem constants (from reference)
// ---------------------------------------------------------------------------
#define B_     2
#define H_     256
#define W_     256
#define C_     256
#define HEADS_ 8
#define CW_    8
#define QK_    32
#define QKC_   256          // HEADS*QK
#define KVC_   512          // QKC + C
#define NWIN_  1024         // (H/CW)*(W/CW)
#define HH_    128          // H/2
#define WW_    128          // W/2
#define L2_EPS_ 1.55e-05f
#define BN_EPS_ 0.001f

typedef __attribute__((ext_vector_type(16))) __bf16 v16bf;
typedef __attribute__((ext_vector_type(2)))  __bf16 v2bf;
typedef __attribute__((ext_vector_type(8)))  float  v8f;

// ---------------------------------------------------------------------------
// WMMA helper (CDNA5 wave32, V_WMMA_F32_16X16X32_BF16)
// ---------------------------------------------------------------------------
static __device__ __forceinline__ v8f wmma_bf16(v16bf a, v16bf b, v8f c) {
  return __builtin_amdgcn_wmma_f32_16x16x32_bf16(false, a, false, b, (short)0, c,
                                                 false, false);
}

// Fragment load from swizzled LDS storage: each lane's 16 bf16 are contiguous
// (32 bytes, 32B aligned) -> 2x ds_load_b128.
static __device__ __forceinline__ v16bf frag_ld(const __bf16* base) {
  return *(const v16bf*)(base + (threadIdx.x & 31) * 16);
}

// B fragment from row-major N x K storage (B[k][n] = src[n][k]); row stride ldr
// must be a multiple of 16 elements and base 32B-aligned so each lane's 16
// contiguous bf16 form an aligned 32-byte load.
static __device__ __forceinline__ v16bf frag_ld_rm(const __bf16* base, int ldr,
                                                   int n0) {
  int lane = threadIdx.x & 31;
  return *(const v16bf*)(base + (size_t)(n0 + (lane & 15)) * ldr +
                         (lane >> 4) * 16);
}

// Swizzle map for the 16x32 bf16 A operand:
//   lane = (m & 15) + ((k >> 4) & 1) * 16
//   vgpr v = (k >> 1) & 7, elem = 2v + (k & 1)
#define A_LANE(m, k) (((m) & 15) + (((k) >> 4) & 1) * 16)
#define A_V(k)       (((k) >> 1) & 7)

// ---------------------------------------------------------------------------
// Kernel 1/5: tiled bf16 WMMA GEMM  Out[M,N] = A[M,K] * W[K,N] (+ bias)
// 256 threads = 8 waves; block tile 128x128; wave tile 32x64; K step 32.
// 8 WMMAs per K-step from 6 fragment loads (2 A + 4 B shared).
// ---------------------------------------------------------------------------
__global__ __launch_bounds__(256) void gemm_bf16_wmma(
    const float* __restrict__ A, const float* __restrict__ Wm,
    const float* __restrict__ bias, int bias_limit, float* __restrict__ Out,
    int M, int N, int K) {
  __shared__ alignas(32) __bf16 As[8 * 32 * 16];  // 8 M-strips of 16 rows, 8KB
  __shared__ alignas(32) __bf16 Bs[8 * 32 * 16];  // 8 N-tiles of 16 cols, 8KB

  const int m_base = blockIdx.x * 128;
  const int n_base = blockIdx.y * 128;
  const int tid = threadIdx.x;
  const int wv  = tid >> 5;
  const int s0  = (wv >> 1) * 2;   // first of two 16-row A strips
  const int t0  = (wv & 1) * 4;    // first of four 16-col B tiles

  v8f acc0[4], acc1[4];
#pragma unroll
  for (int t = 0; t < 4; ++t) { acc0[t] = 0; acc1[t] = 0; }

  for (int k0 = 0; k0 < K; k0 += 32) {
    __syncthreads();
    // prefetch next K-tiles into WGP-scope caches (locality 3 -> WGP prefetch)
    if (k0 + 32 < K) {
      __builtin_prefetch(A + (size_t)(m_base + (tid & 127)) * K + (k0 + 32) +
                             (tid >> 7) * 16, 0, 3);
      __builtin_prefetch(Wm + (size_t)(k0 + 32 + (tid >> 3)) * N + n_base +
                             (tid & 7) * 16, 0, 3);
    }
    // ---- fill A tile 128x32 as swizzled fragments (packed pair stores) ----
#pragma unroll 2
    for (int i = tid; i < 2048; i += 256) {
      int r  = i >> 4;             // 0..127
      int kp = (i & 15) << 1;      // even k
      const float* src = A + (size_t)(m_base + r) * K + k0 + kp;
      v2bf pr = {(__bf16)src[0], (__bf16)src[1]};
      int strip = r >> 4;
      *(v2bf*)&As[((strip * 32 + A_LANE(r, kp)) << 4) + (A_V(kp) << 1)] = pr;
    }
    // ---- fill B tile 32x128: lane=(n&15)+16*(k>>4), elem=k&15 ----
#pragma unroll 2
    for (int i = tid; i < 2048; i += 256) {
      int n  = i & 127;
      int kp = (i >> 7) << 1;      // even k
      const float* src = Wm + (size_t)(k0 + kp) * N + n_base + n;
      v2bf pr = {(__bf16)src[0], (__bf16)src[N]};
      int lane = (n & 15) + ((kp >> 4) & 1) * 16;
      *(v2bf*)&Bs[(((n >> 4) * 32 + lane) << 4) + (kp & 15)] = pr;
    }
    __syncthreads();

    v16bf fa0 = frag_ld(As + s0 * 512);
    v16bf fa1 = frag_ld(As + (s0 + 1) * 512);
#pragma unroll
    for (int t = 0; t < 4; ++t) {
      v16bf fb = frag_ld(Bs + (t0 + t) * 512);
      acc0[t] = wmma_bf16(fa0, fb, acc0[t]);
      acc1[t] = wmma_bf16(fa1, fb, acc1[t]);
    }
  }

  const int lane = tid & 31;
  const int half = lane >> 4;
#pragma unroll
  for (int t = 0; t < 4; ++t) {
    int c = n_base + (t0 + t) * 16 + (lane & 15);
    float bb = (c < bias_limit) ? bias[c] : 0.0f;
#pragma unroll
    for (int v = 0; v < 8; ++v) {
      int mA = m_base + s0 * 16 + v + 8 * half;
      Out[(size_t)mA * N + c]        = acc0[t][v] + bb;
      Out[(size_t)(mA + 16) * N + c] = acc1[t][v] + bb;
    }
  }
}

// ---------------------------------------------------------------------------
// Kernel 2: depthwise 3x3 stride-2 SAME conv on kv channels + BN affine + bias
// qkv: (B,256,256,768), kv = channels 256..767  ->  kvds: (B,128,128,512)
// ---------------------------------------------------------------------------
__global__ __launch_bounds__(256) void dwconv_bn(
    const float* __restrict__ qkv, const float* __restrict__ dw_w,
    const float* __restrict__ gamma, const float* __restrict__ beta,
    const float* __restrict__ v_bias, float* __restrict__ kvds) {
  size_t idx = (size_t)blockIdx.x * 256 + threadIdx.x;  // B*128*128*512
  int c  = idx & 511;
  size_t r = idx >> 9;
  int ox = r & 127; r >>= 7;
  int oy = r & 127; r >>= 7;
  int b  = (int)r;

  float acc = 0.0f;
#pragma unroll
  for (int ky = 0; ky < 3; ++ky) {
    int iy = oy * 2 - 1 + ky;
    if (iy < 0 || iy >= H_) continue;
#pragma unroll
    for (int kx = 0; kx < 3; ++kx) {
      int ix = ox * 2 - 1 + kx;
      if (ix < 0 || ix >= W_) continue;
      float xv = qkv[(((size_t)b * H_ + iy) * W_ + ix) * 768 + 256 + c];
      acc += xv * dw_w[(ky * 3 + kx) * KVC_ + c];
    }
  }
  acc = acc * (gamma[c] * (1.0f / sqrtf(1.0f + BN_EPS_))) + beta[c];
  if (c >= 256) acc += v_bias[c - 256];
  kvds[idx] = acc;
}

// ---------------------------------------------------------------------------
// Kernel 3: CPB MLP -> rbt[225][8] = sigmoid(relu(tab@w1+b1)@w2)*16
// ---------------------------------------------------------------------------
__global__ __launch_bounds__(256) void cpb_kernel(
    const float* __restrict__ w1, const float* __restrict__ b1,
    const float* __restrict__ w2, float* __restrict__ rbt) {
  __shared__ float red[256][8];
  const int i  = blockIdx.x;          // 0..224
  const int iy = i / 15, ix = i % 15;
  const float inv_log8 = 0.48089834696298783f;  // 1/log(8)
  float ty = (float)(iy - 7) * (8.0f / 15.0f);
  float tx = (float)(ix - 7) * (8.0f / 15.0f);
  ty = (ty >= 0.f ? 1.f : -1.f) * log1pf(fabsf(ty)) * inv_log8;
  tx = (tx >= 0.f ? 1.f : -1.f) * log1pf(fabsf(tx)) * inv_log8;

  const int tid = threadIdx.x;
  float part[8];
#pragma unroll
  for (int h = 0; h < 8; ++h) part[h] = 0.0f;
  for (int j = tid; j < 512; j += 256) {
    float hid = fmaxf(ty * w1[j] + tx * w1[512 + j] + b1[j], 0.0f);
#pragma unroll
    for (int h = 0; h < 8; ++h) part[h] += hid * w2[j * 8 + h];
  }
#pragma unroll
  for (int h = 0; h < 8; ++h) red[tid][h] = part[h];
  __syncthreads();
  for (int s = 128; s > 0; s >>= 1) {
    if (tid < s) {
#pragma unroll
      for (int h = 0; h < 8; ++h) red[tid][h] += red[tid + s][h];
    }
    __syncthreads();
  }
  if (tid < 8) rbt[i * 8 + tid] = 16.0f / (1.0f + expf(-red[0][tid]));
}

// ---------------------------------------------------------------------------
// Kernel 4: halo attention.  One 128-thread (4-wave) block per (b, window, h).
// q / attn stored in swizzled A-fragment layout; k / v^T stored row-major with
// 32B-aligned strides so every WMMA fragment is a contiguous 32-byte LDS load.
// ---------------------------------------------------------------------------
__global__ __launch_bounds__(128) void halo_attn(
    const float* __restrict__ qkv, const float* __restrict__ kvds,
    const float* __restrict__ rbt, const float* __restrict__ scale,
    float* __restrict__ obuf) {
  __shared__ alignas(32) __bf16 qs[4 * 32 * 16];       // A frags, 4 strips, 4KB
  __shared__ alignas(32) __bf16 ks[64 * 48];           // row-major NxK, 6KB
  __shared__ alignas(32) __bf16 vT[32 * 80];           // v transposed NxK, 5KB
  __shared__ alignas(32) __bf16 att[2 * 4 * 32 * 16];  // [khalf][strip], 8KB
  __shared__ float  lg[64][66];
  __shared__ float  kmask[64];
  __shared__ float  rbh[225];

  const int blk = blockIdx.x;              // b*8192 + w*8 + h
  const int h  = blk & 7;
  const int w  = (blk >> 3) & 1023;
  const int b  = blk >> 13;
  const int wy = w >> 5, wx = w & 31;
  const int tid  = threadIdx.x;
  const int lane = tid & 31;
  const int wv   = tid >> 5;

  const float t_h = expf(fminf(scale[h], 4.605170185988092f));  // log(100)

  // stage this head's rel-bias slice
  for (int i = tid; i < 225; i += 128) rbh[i] = rbt[i * 8 + h];

  if (tid < 64) {
    // --- q row (bias already applied in GEMM1), l2-norm * t_h ---
    const int r = tid, iy = r >> 3, ix = r & 7;
    const float4* qp = (const float4*)(qkv +
        (((size_t)b * H_ + (wy * 8 + iy)) * W_ + (wx * 8 + ix)) * 768 + h * QK_);
    float buf[QK_];
    float ss = 0.0f;
#pragma unroll
    for (int i = 0; i < 8; ++i) {
      float4 f = qp[i];
      buf[4*i] = f.x; buf[4*i+1] = f.y; buf[4*i+2] = f.z; buf[4*i+3] = f.w;
      ss += f.x*f.x + f.y*f.y + f.z*f.z + f.w*f.w;
    }
    float inv = t_h / sqrtf(fmaxf(ss, L2_EPS_));
    const int strip = r >> 4;
#pragma unroll
    for (int kp = 0; kp < QK_; kp += 2) {
      v2bf pr = {(__bf16)(buf[kp] * inv), (__bf16)(buf[kp + 1] * inv)};
      *(v2bf*)&qs[((strip * 32 + A_LANE(r, kp)) << 4) + (A_V(kp) << 1)] = pr;
    }
  } else {
    // --- k + v halo row ---
    const int r = tid - 64, jy = r >> 3, jx = r & 7;
    const int gy = wy * 4 - 2 + jy;
    const int gx = wx * 4 - 2 + jx;
    const bool inb = (gy >= 0) & (gy < HH_) & (gx >= 0) & (gx < WW_);
    const float4* kp = (const float4*)(kvds +
        (((size_t)b * HH_ + (inb ? gy : 0)) * WW_ + (inb ? gx : 0)) * KVC_ + h * 64);
    float buf[2 * QK_];
    float ss = 0.0f;
#pragma unroll
    for (int i = 0; i < 16; ++i) {
      float4 f = inb ? kp[i] : float4{0.f, 0.f, 0.f, 0.f};
      buf[4*i] = f.x; buf[4*i+1] = f.y; buf[4*i+2] = f.z; buf[4*i+3] = f.w;
      if (i < 8) ss += f.x*f.x + f.y*f.y + f.z*f.z + f.w*f.w;
    }
    float inv = 1.0f / sqrtf(fmaxf(ss, L2_EPS_));
#pragma unroll
    for (int i = 0; i < QK_; i += 2) {
      v2bf pr = {(__bf16)(buf[i] * inv), (__bf16)(buf[i + 1] * inv)};
      *(v2bf*)&ks[r * 48 + i] = pr;
    }
#pragma unroll
    for (int i = 0; i < QK_; ++i) vT[i * 80 + r] = (__bf16)buf[QK_ + i];
    kmask[r] = inb ? 0.0f : -100.0f;
  }
  __syncthreads();

  // --- logits: wave wv owns 16-row M strip, sweeps 4 N tiles ---
  const int m0 = wv * 16;
  const int half = lane >> 4;
  {
    v16bf aq = frag_ld(qs + wv * 512);
#pragma unroll
    for (int nt = 0; nt < 4; ++nt) {
      v16bf bk = frag_ld_rm(ks, 48, nt * 16);
      v8f c = {};
      c = wmma_bf16(aq, bk, c);
      const int n = nt * 16 + (lane & 15);
      const float km = kmask[n];
#pragma unroll
      for (int v = 0; v < 8; ++v) {
        int m = m0 + v + 8 * half;
        int rel = ((m >> 3) - (n >> 3) + 7) * 15 + ((m & 7) - (n & 7) + 7);
        lg[m][n] = c[v] + rbh[rel] + km;
      }
    }
  }
  __syncthreads();

  // --- softmax over 64 keys; write attn in swizzled A-fragment layout ---
  if (tid < 64) {
    float mx = -1e30f;
#pragma unroll
    for (int n = 0; n < 64; ++n) mx = fmaxf(mx, lg[tid][n]);
    float s = 0.0f;
#pragma unroll
    for (int n = 0; n < 64; ++n) {
      float e = expf(lg[tid][n] - mx);
      lg[tid][n] = e;
      s += e;
    }
    float inv = 1.0f / s;
    const int strip = tid >> 4;
#pragma unroll
    for (int k = 0; k < 64; k += 2) {
      int kh = k >> 5, kq = k & 31;
      v2bf pr = {(__bf16)(lg[tid][k] * inv), (__bf16)(lg[tid][k + 1] * inv)};
      *(v2bf*)&att[(((kh * 4 + strip) * 32 + A_LANE(tid, kq)) << 4) +
                   (A_V(kq) << 1)] = pr;
    }
  }
  __syncthreads();

  // --- o = attn(64x64) @ v(64x32): K=64 in two 32-steps ---
  v8f o0 = {}, o1 = {};
#pragma unroll
  for (int kh = 0; kh < 2; ++kh) {
    v16bf aa = frag_ld(att + (kh * 4 + wv) * 512);
    v16bf b0 = frag_ld_rm(vT + kh * 32, 80, 0);
    v16bf b1 = frag_ld_rm(vT + kh * 32, 80, 16);
    o0 = wmma_bf16(aa, b0, o0);
    o1 = wmma_bf16(aa, b1, o1);
  }
  const int n = lane & 15;
#pragma unroll
  for (int v = 0; v < 8; ++v) {
    int m = m0 + v + 8 * half;
    int iy = m >> 3, ix = m & 7;
    size_t base = (((size_t)b * H_ + (wy * 8 + iy)) * W_ + (wx * 8 + ix)) * 256
                  + h * 32;
    obuf[base + n]      = o0[v];
    obuf[base + 16 + n] = o1[v];
  }
}

// ---------------------------------------------------------------------------
// Host-side orchestration
// ---------------------------------------------------------------------------
extern "C" void kernel_launch(void* const* d_in, const int* in_sizes, int n_in,
                              void* d_out, int out_size, void* d_ws,
                              size_t ws_size, hipStream_t stream) {
  (void)in_sizes; (void)n_in; (void)out_size; (void)ws_size;
  const float* x      = (const float*)d_in[0];
  const float* qkv_w  = (const float*)d_in[1];
  const float* dw_w   = (const float*)d_in[2];
  const float* gamma  = (const float*)d_in[3];
  const float* beta   = (const float*)d_in[4];
  const float* q_bias = (const float*)d_in[5];
  const float* v_bias = (const float*)d_in[6];
  const float* scale  = (const float*)d_in[7];
  const float* cpb_w1 = (const float*)d_in[8];
  const float* cpb_b1 = (const float*)d_in[9];
  const float* cpb_w2 = (const float*)d_in[10];
  const float* proj_w = (const float*)d_in[11];
  const float* proj_b = (const float*)d_in[12];
  float* out = (float*)d_out;

  char* ws = (char*)d_ws;
  const size_t qkv_bytes  = (size_t)B_ * H_ * W_ * 768 * sizeof(float);    // 403 MB
  const size_t kvds_bytes = (size_t)B_ * HH_ * WW_ * KVC_ * sizeof(float); //  67 MB
  const size_t obuf_bytes = (size_t)B_ * H_ * W_ * 256 * sizeof(float);    // 134 MB
  float* qkv  = (float*)ws;
  float* kvds = (float*)(ws + qkv_bytes);
  float* obuf = (float*)(ws + qkv_bytes + kvds_bytes);
  float* rbt  = (float*)(ws + qkv_bytes + kvds_bytes + obuf_bytes);

  const int M = B_ * H_ * W_;  // 131072

  // 1) qkv = x @ qkv_w  (+ q_bias on first 256 columns)
  gemm_bf16_wmma<<<dim3(M / 128, 768 / 128), 256, 0, stream>>>(
      x, qkv_w, q_bias, 256, qkv, M, 768, 256);

  // 2) depthwise conv + BN + v_bias
  dwconv_bn<<<(B_ * HH_ * WW_ * KVC_) / 256, 256, 0, stream>>>(
      qkv, dw_w, gamma, beta, v_bias, kvds);

  // 3) continuous-position-bias table
  cpb_kernel<<<225, 256, 0, stream>>>(cpb_w1, cpb_b1, cpb_w2, rbt);

  // 4) halo window attention
  halo_attn<<<B_ * NWIN_ * HEADS_, 128, 0, stream>>>(qkv, kvds, rbt, scale, obuf);

  // 5) out = obuf @ proj_w + proj_b
  gemm_bf16_wmma<<<dim3(M / 128, 256 / 128), 256, 0, stream>>>(
      obuf, proj_w, proj_b, 256, out, M, 256, 256);
}
